// COPY_LAYER_67602785239243
// MI455X (gfx1250) — matrix operand order repaired and checked by reading, not compiled, not run on hardware
//
#include <hip/hip_runtime.h>
#include <hip/hip_bf16.h>

typedef __attribute__((ext_vector_type(16))) __bf16 v16bf;
typedef __attribute__((ext_vector_type(8)))  __bf16 v8bf;
typedef __attribute__((ext_vector_type(8)))  float  v8f;

#define BS 16
#define DEC 100
#define IN_LEN 512
#define HID 768
#define EMB 768
#define VOCAB 32000
#define ROWS (BS*DEC)          // 1600
#define APAD (HID + 8)         // padded LDS row stride in bf16 (1552 B -> bank-conflict free)

// ---------------------------------------------------------------------------
// fp32 -> bf16 conversion of x (row-major 1600 x 768)
// ---------------------------------------------------------------------------
__global__ void conv_x_kernel(const float* __restrict__ x, __bf16* __restrict__ xb, int n) {
    int i = blockIdx.x * blockDim.x + threadIdx.x;
    int stride = gridDim.x * blockDim.x;
    for (; i < n; i += stride) xb[i] = (__bf16)x[i];
}

// ---------------------------------------------------------------------------
// W_vocab (HID x VOCAB, fp32) -> Wt (VOCAB x HID, bf16) via LDS tile transpose
// grid (VOCAB/32, HID/32), block (32, 8)
// ---------------------------------------------------------------------------
__global__ void transpose_w_kernel(const float* __restrict__ W, __bf16* __restrict__ Wt) {
    __shared__ __bf16 tile[32][33];
    int vb = blockIdx.x * 32;
    int kb = blockIdx.y * 32;
    int tx = threadIdx.x, ty = threadIdx.y;
    #pragma unroll
    for (int i = 0; i < 32; i += 8)
        tile[ty + i][tx] = (__bf16)W[(size_t)(kb + ty + i) * VOCAB + vb + tx];
    __syncthreads();
    #pragma unroll
    for (int i = 0; i < 32; i += 8)
        Wt[(size_t)(vb + ty + i) * HID + kb + tx] = tile[tx][ty + i];
}

// ---------------------------------------------------------------------------
// v[b,l] = enc_output[b,l,:] . W_gen[0:HID]   (one wave per row, 8192 rows)
// ---------------------------------------------------------------------------
__global__ void enc_dot_kernel(const float* __restrict__ enc, const float* __restrict__ Wg,
                               float* __restrict__ v) {
    int wave = (blockIdx.x * blockDim.x + threadIdx.x) >> 5;
    int lane = threadIdx.x & 31;
    if (wave >= BS * IN_LEN) return;
    const float* row = enc + (size_t)wave * EMB;
    float s = 0.f;
    for (int i = lane; i < EMB; i += 32) s += row[i] * Wg[i];
    for (int o = 16; o > 0; o >>= 1) s += __shfl_down(s, o, 32);
    if (lane == 0) v[wave] = s;
}

// ---------------------------------------------------------------------------
// p_gen[b,d] = sigmoid( x[b,d,:].Wg[HID:] + attn[b,d,:].v[b,:] + b_gen )
// one wave per (b,d), 1600 waves
// ---------------------------------------------------------------------------
__global__ void pgen_kernel(const float* __restrict__ x, const float* __restrict__ attn,
                            const float* __restrict__ v, const float* __restrict__ Wg,
                            const float* __restrict__ bg, float* __restrict__ pg) {
    int wave = (blockIdx.x * blockDim.x + threadIdx.x) >> 5;
    int lane = threadIdx.x & 31;
    if (wave >= ROWS) return;
    int b = wave / DEC;
    const float* xr = x + (size_t)wave * HID;
    const float* ar = attn + (size_t)wave * IN_LEN;
    const float* vr = v + (size_t)b * IN_LEN;
    float s = 0.f;
    for (int i = lane; i < HID; i += 32) s += xr[i] * Wg[HID + i];
    for (int i = lane; i < IN_LEN; i += 32) s += ar[i] * vr[i];
    for (int o = 16; o > 0; o >>= 1) s += __shfl_down(s, o, 32);
    if (lane == 0) pg[wave] = 1.0f / (1.0f + __expf(-(s + bg[0])));
}

// ---------------------------------------------------------------------------
// WMMA bf16 GEMM: logit = x_bf16(1600x768) * Wt^T(768x32000) + b_vocab
// block = 128 threads (4 waves); A tile (16x768) staged once into LDS via
// CDNA5 async-to-LDS copies (ASYNCcnt), B streamed from L2 with b128 loads.
// wave tile 16Mx64N (4 C frags), block tile 16Mx256N, grid (125, 100)
// ---------------------------------------------------------------------------
__global__ __launch_bounds__(128) void gemm_logits_kernel(
    const __bf16* __restrict__ xb,   // ROWS x HID
    const __bf16* __restrict__ wt,   // VOCAB x HID (transposed W_vocab)
    const float*  __restrict__ bv,   // VOCAB
    float* __restrict__ out)         // ROWS x VOCAB
{
    __shared__ __bf16 As[16 * APAD];             // 16 rows, padded stride

    const int tid  = threadIdx.x;
    const int lane = tid & 31;
    const int wave = tid >> 5;
    const int half = lane >> 4;
    const int l15  = lane & 15;

    const int mbase = blockIdx.y * 16;
    const int nbase = blockIdx.x * 256 + wave * 64;

    // ---- stage A tile: 16*768 bf16 = 24576 B = 1536 x 16B chunks, 12/thread ----
    // global_load_async_to_lds_b128: VDST = VGPR with LDS byte address
    // (low 32 bits of the generic pointer == LDS address per ISA 10.2),
    // VADDR = 64-bit global address. Tracked by ASYNCcnt.
    #pragma unroll
    for (int it = 0; it < 12; ++it) {
        int c    = tid + it * 128;               // chunk id 0..1535
        int row  = c / 96;                       // 96 chunks per 768-bf16 row
        int col  = (c % 96) * 8;                 // bf16 column
        unsigned lds = (unsigned)(size_t)&As[row * APAD + col];
        unsigned long long ga =
            (unsigned long long)(size_t)(xb + (size_t)(mbase + row) * HID + col);
        asm volatile("global_load_async_to_lds_b128 %0, %1, off"
                     :: "v"(lds), "v"(ga) : "memory");
    }
    asm volatile("s_wait_asynccnt 0x0" ::: "memory");
    __syncthreads();

    v8f acc0 = {}, acc1 = {}, acc2 = {}, acc3 = {};

    // A frag from LDS: lane holds row l15; K runs 16*half..16*half+15
    const __bf16* arow = &As[l15 * APAD + 16 * half];
    // B frag: lane holds column (nbase+16f+l15); K runs 16*half..16*half+15
    const __bf16* bptr0 = wt + (size_t)(nbase +  0 + l15) * HID + 16 * half;
    const __bf16* bptr1 = wt + (size_t)(nbase + 16 + l15) * HID + 16 * half;
    const __bf16* bptr2 = wt + (size_t)(nbase + 32 + l15) * HID + 16 * half;
    const __bf16* bptr3 = wt + (size_t)(nbase + 48 + l15) * HID + 16 * half;

    for (int kk = 0; kk < HID; kk += 32) {
        v8bf alo = *(const v8bf*)(arow + kk);        // ds_load_b128
        v8bf ahi = *(const v8bf*)(arow + kk + 8);    // ds_load_b128
        v16bf a;
        #pragma unroll
        for (int i = 0; i < 8; ++i) { a[i] = alo[i]; a[i + 8] = ahi[i]; }

        v16bf b0 = *(const v16bf*)(bptr0 + kk);
        v16bf b1 = *(const v16bf*)(bptr1 + kk);
        v16bf b2 = *(const v16bf*)(bptr2 + kk);
        v16bf b3 = *(const v16bf*)(bptr3 + kk);

        acc0 = __builtin_amdgcn_wmma_f32_16x16x32_bf16(false, a, false, b0, (short)0, acc0, false, false);
        acc1 = __builtin_amdgcn_wmma_f32_16x16x32_bf16(false, a, false, b1, (short)0, acc1, false, false);
        acc2 = __builtin_amdgcn_wmma_f32_16x16x32_bf16(false, a, false, b2, (short)0, acc2, false, false);
        acc3 = __builtin_amdgcn_wmma_f32_16x16x32_bf16(false, a, false, b3, (short)0, acc3, false, false);
    }

    v8f accs[4] = {acc0, acc1, acc2, acc3};
    #pragma unroll
    for (int f = 0; f < 4; ++f) {
        int col = nbase + f * 16 + l15;
        float bias = bv[col];
        #pragma unroll
        for (int g = 0; g < 8; ++g) {
            int row = mbase + g + 8 * half;   // C/D layout: VGPR g -> M = g + 8*half
            out[(size_t)row * VOCAB + col] = accs[f][g] + bias;
        }
    }
}

// ---------------------------------------------------------------------------
// block reductions (256 threads = 8 waves)
// ---------------------------------------------------------------------------
__device__ __forceinline__ float block_reduce_max(float v, float* red) {
    for (int o = 16; o > 0; o >>= 1) v = fmaxf(v, __shfl_down(v, o, 32));
    int lane = threadIdx.x & 31, w = threadIdx.x >> 5, nw = blockDim.x >> 5;
    if (lane == 0) red[w] = v;
    __syncthreads();
    if (w == 0) {
        float t = (lane < nw) ? red[lane] : -3.4e38f;
        for (int o = 16; o > 0; o >>= 1) t = fmaxf(t, __shfl_down(t, o, 32));
        if (lane == 0) red[0] = t;
    }
    __syncthreads();
    float r = red[0];
    __syncthreads();
    return r;
}

__device__ __forceinline__ float block_reduce_sum(float v, float* red) {
    for (int o = 16; o > 0; o >>= 1) v += __shfl_down(v, o, 32);
    int lane = threadIdx.x & 31, w = threadIdx.x >> 5, nw = blockDim.x >> 5;
    if (lane == 0) red[w] = v;
    __syncthreads();
    if (w == 0) {
        float t = (lane < nw) ? red[lane] : 0.f;
        for (int o = 16; o > 0; o >>= 1) t += __shfl_down(t, o, 32);
        if (lane == 0) red[0] = t;
    }
    __syncthreads();
    float r = red[0];
    __syncthreads();
    return r;
}

// ---------------------------------------------------------------------------
// in-place: out[row,:] = softmax(out[row,:]) * p_gen[row]; one block per row
// ---------------------------------------------------------------------------
__global__ __launch_bounds__(256) void vocab_softmax_kernel(float* __restrict__ out,
                                                            const float* __restrict__ pg) {
    int row = blockIdx.x;
    float* r = out + (size_t)row * VOCAB;
    __shared__ float red[8];
    float m = -3.4e38f;
    for (int i = threadIdx.x; i < VOCAB; i += 256) m = fmaxf(m, r[i]);
    m = block_reduce_max(m, red);
    float s = 0.f;
    for (int i = threadIdx.x; i < VOCAB; i += 256) s += __expf(r[i] - m);
    s = block_reduce_sum(s, red);
    float scale = pg[row] / s;
    for (int i = threadIdx.x; i < VOCAB; i += 256) r[i] = __expf(r[i] - m) * scale;
}

// ---------------------------------------------------------------------------
// fused attn softmax + scatter-add: out[row, enc_input[b,l]] += softmax(attn)[l]*(1-pg)
// one block (256 thr) per (b,d) row of 512
// ---------------------------------------------------------------------------
__global__ __launch_bounds__(256) void attn_scatter_kernel(const float* __restrict__ attn,
                                                           const int* __restrict__ enc_in,
                                                           const float* __restrict__ pg,
                                                           float* __restrict__ out) {
    int row = blockIdx.x;          // b*DEC + d
    int b = row / DEC;
    const float* ar = attn + (size_t)row * IN_LEN;
    const int* ei = enc_in + (size_t)b * IN_LEN;
    float* orow = out + (size_t)row * VOCAB;
    __shared__ float red[8];
    int tid = threadIdx.x;
    float a0 = ar[tid], a1 = ar[tid + 256];
    float m = block_reduce_max(fmaxf(a0, a1), red);
    float e0 = __expf(a0 - m), e1 = __expf(a1 - m);
    float s = block_reduce_sum(e0 + e1, red);
    float scale = (1.0f - pg[row]) / s;
    unsafeAtomicAdd(orow + ei[tid], e0 * scale);
    unsafeAtomicAdd(orow + ei[tid + 256], e1 * scale);
}

// ---------------------------------------------------------------------------
// elementwise log
// ---------------------------------------------------------------------------
__global__ void log_kernel(float* __restrict__ out, int n) {
    int i = blockIdx.x * blockDim.x + threadIdx.x;
    int stride = gridDim.x * blockDim.x;
    for (; i < n; i += stride) out[i] = __logf(out[i]);
}

// ---------------------------------------------------------------------------
extern "C" void kernel_launch(void* const* d_in, const int* in_sizes, int n_in,
                              void* d_out, int out_size, void* d_ws, size_t ws_size,
                              hipStream_t stream) {
    const float* x          = (const float*)d_in[0];
    const float* attn_dist  = (const float*)d_in[1];
    const int*   enc_input  = (const int*)  d_in[2];
    const float* enc_output = (const float*)d_in[3];
    const float* W_vocab    = (const float*)d_in[4];
    const float* b_vocab    = (const float*)d_in[5];
    const float* W_gen      = (const float*)d_in[6];
    const float* b_gen      = (const float*)d_in[7];
    float* out = (float*)d_out;

    // workspace carve-up (all offsets 256B aligned)
    char* ws = (char*)d_ws;
    __bf16* Wt = (__bf16*)ws;                                   // VOCAB*HID*2   = 49,152,000 B
    __bf16* xb = (__bf16*)(ws + (size_t)VOCAB * HID * 2);       // ROWS*HID*2    =  2,457,600 B
    float*  vbl = (float*)(ws + (size_t)VOCAB * HID * 2 + (size_t)ROWS * HID * 2);          // 32,768 B
    float*  pg  = (float*)(ws + (size_t)VOCAB * HID * 2 + (size_t)ROWS * HID * 2 + 32768);  //  6,400 B

    conv_x_kernel<<<1200, 256, 0, stream>>>(x, xb, ROWS * HID);
    transpose_w_kernel<<<dim3(VOCAB / 32, HID / 32), dim3(32, 8), 0, stream>>>(W_vocab, Wt);
    enc_dot_kernel<<<(BS * IN_LEN) / 8, 256, 0, stream>>>(enc_output, W_gen, vbl);
    pgen_kernel<<<ROWS / 8, 256, 0, stream>>>(x, attn_dist, vbl, W_gen, b_gen, pg);
    gemm_logits_kernel<<<dim3(VOCAB / 256, ROWS / 16), 128, 0, stream>>>(xb, Wt, b_vocab, out);
    vocab_softmax_kernel<<<ROWS, 256, 0, stream>>>(out, pg);
    attn_scatter_kernel<<<ROWS, 256, 0, stream>>>(attn_dist, enc_input, pg, out);
    log_kernel<<<2048, 256, 0, stream>>>(out, ROWS * VOCAB);
}
